// HybridMemory_20074677141926
// MI455X (gfx1250) — matrix operand order repaired and checked by reading, not compiled, not run on hardware
//
#include <hip/hip_runtime.h>
#include <hip/hip_bf16.h>

#define B_SZ   2048
#define S_SZ   55260
#define D_SZ   2048
#define C_SZ   1000
#define C_PAD  1024      // 64 * 16 tiles; pad clusters have counts==0 -> masked out
#define TEMP   0.05f
#define EPS    1e-6f

typedef __attribute__((ext_vector_type(16))) __bf16        v16bf;
typedef __attribute__((ext_vector_type(8)))  float         v8f;
typedef __attribute__((ext_vector_type(4)))  unsigned int  v4u;

union FragU { v4u u[2]; v16bf v; };

__device__ __forceinline__ unsigned short f32_to_bf16(float f) {
    unsigned int u = __float_as_uint(f);
    u += 0x7fffu + ((u >> 16) & 1u);      // round to nearest even
    return (unsigned short)(u >> 16);
}

// ---------------- 1. cluster histogram -> counts[C_PAD] (float) ----------------
__global__ void count_kernel(const int* __restrict__ labels, float* __restrict__ counts) {
    __shared__ unsigned int h[C_PAD];
    for (int i = threadIdx.x; i < C_PAD; i += blockDim.x) h[i] = 0u;
    __syncthreads();
    for (int s = threadIdx.x; s < S_SZ; s += blockDim.x)
        atomicAdd(&h[labels[s]], 1u);
    __syncthreads();
    for (int i = threadIdx.x; i < C_PAD; i += blockDim.x)
        counts[i] = (float)h[i];
}

// ---------------- 2. inputs f32 -> bf16 ----------------
__global__ void cvt_kernel(const float* __restrict__ in, unsigned short* __restrict__ out, int n) {
    int i = blockIdx.x * blockDim.x + threadIdx.x;
    if (i < n) out[i] = f32_to_bf16(in[i]);
}

// ---------------- 3. centroids: segment-sum of features by label ----------------
// One workgroup per 32-column tile. 125 KB dynamic LDS: acc[C_SZ][32] (f32).
// Wave reads 128 contiguous bytes of a feature row; lane l accumulates column l
// into acc[label*32 + l] -> bank-conflict-free ds_add_f32.
__global__ void centroid_kernel(const float* __restrict__ features,
                                const int*   __restrict__ labels,
                                unsigned short* __restrict__ cent) {
    extern __shared__ float acc[];                 // C_SZ * 32 floats
    const int tid  = threadIdx.x;
    const int lane = tid & 31;
    const int wave = tid >> 5;                     // 8 waves / 256 threads
    const int col0 = blockIdx.x * 32;

    for (int i = tid; i < C_SZ * 32; i += blockDim.x) acc[i] = 0.0f;
    __syncthreads();

    for (int s = wave; s < S_SZ; s += 8) {
        if (s + 64 < S_SZ)                         // keep the HBM stream deep
            __builtin_prefetch(&features[(size_t)(s + 64) * D_SZ + col0 + lane], 0, 1);
        float v = features[(size_t)s * D_SZ + col0 + lane];
        int   c = labels[s];
        atomicAdd(&acc[c * 32 + lane], v);
    }
    __syncthreads();

    for (int i = tid; i < C_PAD * 32; i += blockDim.x) {
        int c = i >> 5, col = i & 31;
        float v = (c < C_SZ) ? acc[i] : 0.0f;      // pad clusters -> zero rows
        cent[(size_t)c * D_SZ + col0 + col] = f32_to_bf16(v);
    }
}

// ---------------- 4. WMMA GEMM: logits[B][C_PAD] = (inputs @ cent^T) / (TEMP*nums) ----------------
// Register-blocked: one wave computes a 64(M) x 32(N) region = 4x2 WMMA tiles.
// Only TWO live 64-bit base pointers; tile row offsets (j*65536 B, i*65536 B) and
// the k-advance are compile-time constants folded into the 24-bit load IOFFSET,
// so nothing spills and 8 independent WMMA accumulator chains stay in flight.
__global__ void gemm_kernel(const unsigned short* __restrict__ A,   // [B_SZ, D_SZ] bf16
                            const unsigned short* __restrict__ Bc,  // [C_PAD, D_SZ] bf16
                            const float* __restrict__ counts,
                            float* __restrict__ logits) {
    const int lane  = threadIdx.x & 31;
    const int half  = lane >> 4;          // 0: lanes 0-15, 1: lanes 16-31
    const int idx16 = lane & 15;
    const int nblk  = blockIdx.x;         // 0..31 (32 cols each)
    const int mblk  = blockIdx.y;         // 0..31 (64 rows each)

    // A layout (16-bit 16x32): lanes 0-15 hold K {0-7,16-23}; lanes 16-31 hold K {8-15,24-31}
    const unsigned short* ap = A  + (size_t)(mblk * 64 + idx16) * D_SZ + half * 8;
    // B layout (16-bit 32x16): lanes 0-15 hold K {0-7,16-23} of col n; lanes 16-31 hold K {8-15,24-31}
    const unsigned short* bp = Bc + (size_t)(nblk * 32 + idx16) * D_SZ + half * 8;

    v8f acc[4][2] = {};
    for (int k = 0; k < D_SZ; k += 32) {
        FragU fa[4], fb[2];
#pragma unroll
        for (int j = 0; j < 4; ++j) {
            const unsigned short* p = ap + j * (16 * D_SZ) + k;   // +j*65536 B (immediate)
            fa[j].u[0] = *(const v4u*)(p);
            fa[j].u[1] = *(const v4u*)(p + 16);
        }
#pragma unroll
        for (int i = 0; i < 2; ++i) {
            const unsigned short* p = bp + i * (16 * D_SZ) + k;   // +i*65536 B (immediate)
            fb[i].u[0] = *(const v4u*)(p);
            fb[i].u[1] = *(const v4u*)(p + 16);
        }
#pragma unroll
        for (int j = 0; j < 4; ++j)
#pragma unroll
            for (int i = 0; i < 2; ++i)
                acc[j][i] = __builtin_amdgcn_wmma_f32_16x16x32_bf16(
                                false, fa[j].v, false, fb[i].v,
                                (short)0, acc[j][i], false, false);
    }

#pragma unroll
    for (int i = 0; i < 2; ++i) {
        const int   col  = nblk * 32 + i * 16 + idx16;
        const float nums = counts[col];
        const float sc   = (nums > 0.0f) ? (1.0f / (TEMP * nums)) : 0.0f;
#pragma unroll
        for (int j = 0; j < 4; ++j) {
            // C/D layout: VGPR r holds M = base + r + half*8, N = col
            const int rbase = mblk * 64 + j * 16 + half * 8;
#pragma unroll
            for (int r = 0; r < 8; ++r)
                logits[(size_t)(rbase + r) * C_PAD + col] = acc[j][i][r] * sc;
        }
    }
}

// ---------------- 5. per-row masked-softmax NLL ----------------
__global__ void rowloss_kernel(const float* __restrict__ logits,
                               const float* __restrict__ counts,
                               const int*   __restrict__ labels,
                               const int*   __restrict__ pids,
                               float* __restrict__ rowloss) {
    const int row = blockIdx.x;
    const int t   = labels[pids[row]];    // target cluster (always non-empty)
    __shared__ float red[256];
    __shared__ float s_et;

    float sum = 0.0f;
    for (int c = threadIdx.x; c < C_PAD; c += 256) {
        float v = logits[(size_t)row * C_PAD + c];
        float e = (counts[c] > 0.0f) ? expf(v) : 0.0f;
        sum += e;
        if (c == t) s_et = e;             // exactly one thread writes
    }
    red[threadIdx.x] = sum;
    __syncthreads();
    for (int o = 128; o > 0; o >>= 1) {
        if (threadIdx.x < o) red[threadIdx.x] += red[threadIdx.x + o];
        __syncthreads();
    }
    if (threadIdx.x == 0) {
        float prob = s_et / (red[0] + EPS);
        rowloss[row] = -logf(prob + EPS);
    }
}

// ---------------- 6. deterministic final mean ----------------
__global__ void finalize_kernel(const float* __restrict__ rowloss, float* __restrict__ out) {
    __shared__ float red[256];
    float s = 0.0f;
    for (int i = threadIdx.x; i < B_SZ; i += 256) s += rowloss[i];
    red[threadIdx.x] = s;
    __syncthreads();
    for (int o = 128; o > 0; o >>= 1) {
        if (threadIdx.x < o) red[threadIdx.x] += red[threadIdx.x + o];
        __syncthreads();
    }
    if (threadIdx.x == 0) out[0] = red[0] / (float)B_SZ;
}

extern "C" void kernel_launch(void* const* d_in, const int* in_sizes, int n_in,
                              void* d_out, int out_size, void* d_ws, size_t ws_size,
                              hipStream_t stream) {
    (void)in_sizes; (void)n_in; (void)out_size; (void)ws_size;

    const float* inputs   = (const float*)d_in[0];   // [B_SZ, D_SZ]
    const float* features = (const float*)d_in[1];   // [S_SZ, D_SZ]
    const int*   labels   = (const int*)d_in[2];     // [S_SZ]
    const int*   pids     = (const int*)d_in[3];     // [B_SZ]
    float*       out      = (float*)d_out;           // scalar

    // workspace layout (256B aligned)
    char* w = (char*)d_ws;
    float*          counts  = (float*)w;                                    // 1024*4 = 4,096
    unsigned short* inb     = (unsigned short*)(w + 4096);                  // B*D*2  = 8,388,608
    unsigned short* cent    = (unsigned short*)(w + 4096 + 8388608);        // C_PAD*D*2 = 4,194,304
    float*          logits  = (float*)(w + 4096 + 8388608 + 4194304);       // B*C_PAD*4 = 8,388,608
    float*          rowloss = (float*)(w + 4096 + 8388608 + 4194304 + 8388608); // 8,192

    // 1. cluster sizes
    count_kernel<<<1, 1024, 0, stream>>>(labels, counts);

    // 2. inputs -> bf16
    {
        int n = B_SZ * D_SZ;
        cvt_kernel<<<(n + 255) / 256, 256, 0, stream>>>(inputs, inb, n);
    }

    // 3. centroids (dynamic LDS: 1000 clusters x 32 cols x f32 = 128,000 B)
    centroid_kernel<<<D_SZ / 32, 256, C_SZ * 32 * sizeof(float), stream>>>(features, labels, cent);

    // 4. WMMA GEMM + fused temperature / cluster-mean scaling (4x2 tiles per wave)
    gemm_kernel<<<dim3(C_PAD / 32, B_SZ / 64), 32, 0, stream>>>(inb, cent, counts, logits);

    // 5. masked softmax + NLL per row
    rowloss_kernel<<<B_SZ, 256, 0, stream>>>(logits, counts, labels, pids, rowloss);

    // 6. mean
    finalize_kernel<<<1, 256, 0, stream>>>(rowloss, out);
}
